// LstmModel_31851477467289
// MI455X (gfx1250) — compile-verified, matrix operand order checked by reading, not run on hardware
//
#include <hip/hip_runtime.h>

#define T_STEPS 512
#define BATCH   256
#define HID     256
#define NSTATE  (BATCH*HID)   // 65536
#define NBLOCKS 128
#define SLAB    4096          // u16 per 16-col weight slab (8 kc * 32 lanes * 16)

typedef __attribute__((ext_vector_type(16))) __bf16       v16bf;
typedef __attribute__((ext_vector_type(8)))  float        v8f;
typedef __attribute__((ext_vector_type(4)))  unsigned int u32x4;

union Frag { v16bf v; unsigned short u[16]; u32x4 q[2]; };

__device__ __forceinline__ unsigned short f2bf(float f) {
  unsigned int x = __float_as_uint(f);
  return (unsigned short)((x + 0x7fffu + ((x >> 16) & 1u)) >> 16);   // RNE
}
__device__ __forceinline__ float sigm(float x) { return 1.0f / (1.0f + __expf(-x)); }

// Three gate GEMM slabs fused over one A-fragment stream; B from LDS.
// A: bf16 row-major state [256x256] (global, changes per step) -> 2x b128/chunk.
// B: LDS slabs, split-half layout (q0 at +0, q1 at +2048 u16) -> conflict-minimal ds_load_b128.
__device__ __forceinline__ void mm3_lds(v8f& a0, v8f& a1, v8f& a2,
                                        const unsigned short* __restrict__ S, int m0,
                                        const unsigned short* sm, int lane) {
  const int mrow = m0 + (lane & 15);
  const int kb8  = (lane >> 4) << 3;
  const unsigned short* srow = S + (size_t)mrow * HID;
#pragma unroll
  for (int kc = 0; kc < 8; ++kc) {
    Frag a, b0, b1, b2;
    const u32x4* ap = (const u32x4*)(srow + kc * 32 + kb8);
    a.q[0] = ap[0];                               // K +0..7
    a.q[1] = ap[2];                               // K +16..23
    const int li = (kc * 32 + lane) * 8;          // u16 index within half-slab
    b0.q[0] = *(const u32x4*)(sm + li);
    b0.q[1] = *(const u32x4*)(sm + 2048 + li);
    b1.q[0] = *(const u32x4*)(sm + SLAB + li);
    b1.q[1] = *(const u32x4*)(sm + SLAB + 2048 + li);
    b2.q[0] = *(const u32x4*)(sm + 2 * SLAB + li);
    b2.q[1] = *(const u32x4*)(sm + 2 * SLAB + 2048 + li);
    a0 = __builtin_amdgcn_wmma_f32_16x16x32_bf16(false, a.v, false, b0.v, (short)0, a0, false, false);
    a1 = __builtin_amdgcn_wmma_f32_16x16x32_bf16(false, a.v, false, b1.v, (short)0, a1, false, false);
    a2 = __builtin_amdgcn_wmma_f32_16x16x32_bf16(false, a.v, false, b2.v, (short)0, a2, false, false);
  }
}

// Single-slab accumulate (plin), B from LDS slab 0.
__device__ __forceinline__ void mm_lds(v8f& acc, const unsigned short* __restrict__ S, int m0,
                                       const unsigned short* sm, int lane) {
  const int mrow = m0 + (lane & 15);
  const int kb8  = (lane >> 4) << 3;
  const unsigned short* srow = S + (size_t)mrow * HID;
#pragma unroll
  for (int kc = 0; kc < 8; ++kc) {
    Frag a, b0;
    const u32x4* ap = (const u32x4*)(srow + kc * 32 + kb8);
    a.q[0] = ap[0];
    a.q[1] = ap[2];
    const int li = (kc * 32 + lane) * 8;
    b0.q[0] = *(const u32x4*)(sm + li);
    b0.q[1] = *(const u32x4*)(sm + 2048 + li);
    acc = __builtin_amdgcn_wmma_f32_16x16x32_bf16(false, a.v, false, b0.v, (short)0, acc, false, false);
  }
}

// Block-cooperative copy of one 8KB weight slab global->LDS.
__device__ __forceinline__ void stage_slab(unsigned short* dst, const unsigned short* src) {
  const u32x4* s4 = (const u32x4*)src;
  u32x4* d4 = (u32x4*)dst;
#pragma unroll
  for (int i = 0; i < 2; ++i) d4[threadIdx.x + 256 * i] = s4[threadIdx.x + 256 * i]; // 512 x 16B
}

__device__ __forceinline__ void grid_sync(unsigned* cnt, unsigned* gen, unsigned nb) {
  __syncthreads();
  if (threadIdx.x == 0) {
    __threadfence();
    unsigned g = __hip_atomic_load(gen, __ATOMIC_RELAXED, __HIP_MEMORY_SCOPE_AGENT);
    unsigned a = __hip_atomic_fetch_add(cnt, 1u, __ATOMIC_ACQ_REL, __HIP_MEMORY_SCOPE_AGENT);
    if (a == nb - 1u) {
      __hip_atomic_store(cnt, 0u, __ATOMIC_RELAXED, __HIP_MEMORY_SCOPE_AGENT);
      __hip_atomic_fetch_add(gen, 1u, __ATOMIC_RELEASE, __HIP_MEMORY_SCOPE_AGENT);
    } else {
      while (__hip_atomic_load(gen, __ATOMIC_ACQUIRE, __HIP_MEMORY_SCOPE_AGENT) == g)
        __builtin_amdgcn_s_sleep(2);
    }
    __threadfence();
  }
  __syncthreads();
}

// ---- prep kernels -----------------------------------------------------------
// Packed slab layout per 16-output-col tile tjj (SLAB=4096 u16):
//   q0 half: base + (kc*32+lane)*8 + e        -> K = kc*32 + (lane>>4)*8 + e
//   q1 half: base + 2048 + (kc*32+lane)*8 + e -> K = ... + 16 + e
__global__ void pack_weights(const float* __restrict__ src, unsigned short* __restrict__ dst, int R) {
  int idx = blockIdx.x * blockDim.x + threadIdx.x;          // (tjj*8+kc)*32+lane
  if (idx >= (R >> 4) * 8 * 32) return;
  int lane = idx & 31, kc = (idx >> 5) & 7, tjj = idx >> 8;
  int n  = tjj * 16 + (lane & 15);
  int kb = kc * 32 + ((lane >> 4) << 3);
  unsigned short* out0 = dst + (size_t)tjj * SLAB + (size_t)(idx & 255) * 8;
  unsigned short* out1 = out0 + 2048;
  const float* s = src + (size_t)n * HID + kb;
#pragma unroll
  for (int e = 0; e < 8; ++e) { out0[e] = f2bf(s[e]); out1[e] = f2bf(s[16 + e]); }
}

__global__ void init_states(const float* __restrict__ he0, const float* __restrict__ hm0,
                            const float* __restrict__ hp0,
                            unsigned short* __restrict__ He, unsigned short* __restrict__ Hm,
                            unsigned short* __restrict__ Hp) {
  int i = blockIdx.x * blockDim.x + threadIdx.x;
  if (i < NSTATE) {                    // t = -1 lives in buffer 1  ((-1)&1 == 1)
    He[NSTATE + i] = f2bf(he0[i]);
    Hm[NSTATE + i] = f2bf(hm0[i]);
    Hp[NSTATE + i] = f2bf(hp0[i]);
  }
}

__global__ void init_mask(float* __restrict__ mask, const float* __restrict__ mlin_b) {
  int i = blockIdx.x * blockDim.x + threadIdx.x;
  if (i < T_STEPS * BATCH) mask[i] = mlin_b[0];
}

// ---- persistent recurrent kernel -------------------------------------------
__global__ void __launch_bounds__(256) rnn_persistent(
    const float* __restrict__ seq, const float* __restrict__ delay,
    const float* __restrict__ enc_w_ih, const float* __restrict__ enc_b_ih,
    const float* __restrict__ enc_b_hh,
    const float* __restrict__ mdec_b_ih, const float* __restrict__ mdec_b_hh,
    const float* __restrict__ mlin_w,
    const float* __restrict__ pdec_b_ih, const float* __restrict__ pdec_b_hh,
    const float* __restrict__ plin_b,
    const float* __restrict__ h0e, const float* __restrict__ h0m, const float* __restrict__ h0p,
    const unsigned short* __restrict__ Wenc_hh,
    const unsigned short* __restrict__ Wm_ih, const unsigned short* __restrict__ Wm_hh,
    const unsigned short* __restrict__ Wp_ih, const unsigned short* __restrict__ Wp_hh,
    const unsigned short* __restrict__ Wplin,
    unsigned short* __restrict__ He, unsigned short* __restrict__ Hm,
    unsigned short* __restrict__ Hp,
    float* __restrict__ mask_out, float* __restrict__ pred_out,
    unsigned* sync_cnt, unsigned* sync_gen)
{
  const int lane = threadIdx.x & 31;
  const int gw   = blockIdx.x * 8 + (threadIdx.x >> 5);     // 0..1023
  const int kind = gw >> 8;                                 // 0 enc, 1 mdec, 2 pdec, 3 plin
  const int tid  = gw & 255;
  const int m0 = (tid & 15) * 16;                           // row tile: varies inside block
  const int tj = tid >> 4;                                  // col tile: uniform per block
  const int oo = tj * 16 + (lane & 15);                     // output column of this lane
  const int bofs = (lane >> 4) << 3;                        // C-layout row offset

  // Stage this block's (block-uniform) weight slabs into LDS once: <=48KB.
  __shared__ unsigned short smem[6 * SLAB];
  if (kind == 0) {
    stage_slab(smem,            Wenc_hh + (size_t)(tj     ) * SLAB);
    stage_slab(smem + SLAB,     Wenc_hh + (size_t)(tj + 16) * SLAB);
    stage_slab(smem + 2 * SLAB, Wenc_hh + (size_t)(tj + 32) * SLAB);
  } else if (kind == 1) {
    stage_slab(smem,            Wm_ih + (size_t)(tj     ) * SLAB);
    stage_slab(smem + SLAB,     Wm_ih + (size_t)(tj + 16) * SLAB);
    stage_slab(smem + 2 * SLAB, Wm_ih + (size_t)(tj + 32) * SLAB);
    stage_slab(smem + 3 * SLAB, Wm_hh + (size_t)(tj     ) * SLAB);
    stage_slab(smem + 4 * SLAB, Wm_hh + (size_t)(tj + 16) * SLAB);
    stage_slab(smem + 5 * SLAB, Wm_hh + (size_t)(tj + 32) * SLAB);
  } else if (kind == 2) {
    stage_slab(smem,            Wp_ih + (size_t)(tj     ) * SLAB);
    stage_slab(smem + SLAB,     Wp_ih + (size_t)(tj + 16) * SLAB);
    stage_slab(smem + 2 * SLAB, Wp_ih + (size_t)(tj + 32) * SLAB);
    stage_slab(smem + 3 * SLAB, Wp_hh + (size_t)(tj     ) * SLAB);
    stage_slab(smem + 4 * SLAB, Wp_hh + (size_t)(tj + 16) * SLAB);
    stage_slab(smem + 5 * SLAB, Wp_hh + (size_t)(tj + 32) * SLAB);
  } else {
    stage_slab(smem,            Wplin + (size_t)tj * SLAB);
  }
  __syncthreads();

  // Register-resident fp32 state: this wave owns tile (m0, tj) of its stack's
  // hidden state for the entire sequence -> z*h term never touches memory.
  v8f hold = {};
  if (kind == 0) {
#pragma unroll
    for (int v = 0; v < 8; ++v) hold[v] = h0e[(size_t)(m0 + v + bofs) * HID + oo];
  } else if (kind == 1) {
#pragma unroll
    for (int v = 0; v < 8; ++v) hold[v] = h0m[(size_t)(m0 + v + bofs) * HID + oo];
  } else if (kind == 2) {
#pragma unroll
    for (int v = 0; v < 8; ++v) hold[v] = h0p[(size_t)(m0 + v + bofs) * HID + oo];
  }

  v8f acc_pred = {};                                        // plin running mean, registers all T steps

  for (int it = 0; it <= T_STEPS; ++it) {
    if (kind == 0) {                                        // enc: h_e(it)
      if (it < T_STEPS) {
        const int t = it;
        const unsigned short* hprev = He + (size_t)((t - 1) & 1) * NSTATE;
        unsigned short*       hnext = He + (size_t)(t & 1) * NSTATE;
        v8f ar = {}, az = {}, an = {};
        mm3_lds(ar, az, an, hprev, m0, smem, lane);
        const float wr = enc_w_ih[oo], wz = enc_w_ih[HID + oo], wn = enc_w_ih[2 * HID + oo];
        const float cr = enc_b_ih[oo] + enc_b_hh[oo];
        const float cz = enc_b_ih[HID + oo] + enc_b_hh[HID + oo];
        const float bin_ = enc_b_ih[2 * HID + oo];
        const float bhn  = enc_b_hh[2 * HID + oo];
#pragma unroll
        for (int v = 0; v < 8; ++v) {
          const int b = m0 + v + bofs;
          const float x = (t == 0) ? delay[b] : seq[(size_t)(t - 1) * BATCH + b];
          const float r = sigm(x * wr + ar[v] + cr);
          const float z = sigm(x * wz + az[v] + cz);
          const float n = tanhf(x * wn + bin_ + r * (an[v] + bhn));
          const float hnew = (1.0f - z) * n + z * hold[v];
          hold[v] = hnew;
          hnext[(size_t)b * HID + oo] = f2bf(hnew);
        }
      }
    } else if (kind == 1) {                                 // mdec: h_m(it-1) + fused mask
      if (it >= 1) {
        const int t = it - 1;
        const unsigned short* he     = He + (size_t)(t & 1) * NSTATE;
        const unsigned short* hmprev = Hm + (size_t)((t - 1) & 1) * NSTATE;
        unsigned short*       hmnext = Hm + (size_t)(t & 1) * NSTATE;
        v8f ar = {}, az = {}, ain = {}, ahn = {};
        mm3_lds(ar, az, ain, he,     m0, smem,            lane);
        mm3_lds(ar, az, ahn, hmprev, m0, smem + 3 * SLAB, lane);
        const float cr = mdec_b_ih[oo] + mdec_b_hh[oo];
        const float cz = mdec_b_ih[HID + oo] + mdec_b_hh[HID + oo];
        const float bin_ = mdec_b_ih[2 * HID + oo];
        const float bhn  = mdec_b_hh[2 * HID + oo];
        const float wm = mlin_w[oo];
#pragma unroll
        for (int v = 0; v < 8; ++v) {
          const int b = m0 + v + bofs;
          const float r = sigm(ar[v] + cr);
          const float z = sigm(az[v] + cz);
          const float n = tanhf(ain[v] + bin_ + r * (ahn[v] + bhn));
          const float hnew = (1.0f - z) * n + z * hold[v];
          hold[v] = hnew;
          hmnext[(size_t)b * HID + oo] = f2bf(hnew);
          float part = hnew * wm;                           // mask_t[b] partial (F==1)
          part += __shfl_xor(part, 1);
          part += __shfl_xor(part, 2);
          part += __shfl_xor(part, 4);
          part += __shfl_xor(part, 8);
          if ((lane & 15) == 0) atomicAdd(&mask_out[(size_t)t * BATCH + b], part);
        }
      }
    } else if (kind == 2) {                                 // pdec: h_p(it-1)
      if (it >= 1) {
        const int t = it - 1;
        const unsigned short* he     = He + (size_t)(t & 1) * NSTATE;
        const unsigned short* hpprev = Hp + (size_t)((t - 1) & 1) * NSTATE;
        unsigned short*       hpnext = Hp + (size_t)(t & 1) * NSTATE;
        v8f ar = {}, az = {}, ain = {}, ahn = {};
        mm3_lds(ar, az, ain, he,     m0, smem,            lane);
        mm3_lds(ar, az, ahn, hpprev, m0, smem + 3 * SLAB, lane);
        const float cr = pdec_b_ih[oo] + pdec_b_hh[oo];
        const float cz = pdec_b_ih[HID + oo] + pdec_b_hh[HID + oo];
        const float bin_ = pdec_b_ih[2 * HID + oo];
        const float bhn  = pdec_b_hh[2 * HID + oo];
#pragma unroll
        for (int v = 0; v < 8; ++v) {
          const int b = m0 + v + bofs;
          const float r = sigm(ar[v] + cr);
          const float z = sigm(az[v] + cz);
          const float n = tanhf(ain[v] + bin_ + r * (ahn[v] + bhn));
          const float hnew = (1.0f - z) * n + z * hold[v];
          hold[v] = hnew;
          hpnext[(size_t)b * HID + oo] = f2bf(hnew);
        }
      }
    } else {                                                // plin: accumulate h_p(it-2) @ plin_w^T
      if (it >= 2) {
        const int t = it - 2;
        mm_lds(acc_pred, Hp + (size_t)(t & 1) * NSTATE, m0, smem, lane);
      }
    }
    grid_sync(sync_cnt, sync_gen, NBLOCKS);                 // single device barrier per step
  }

  if (kind == 3) {                                          // epilogue: last h_p + mean + bias
    mm_lds(acc_pred, Hp + (size_t)((T_STEPS - 1) & 1) * NSTATE, m0, smem, lane);
    const float bias = plin_b[oo];
#pragma unroll
    for (int v = 0; v < 8; ++v) {
      const int b = m0 + v + bofs;
      pred_out[(size_t)b * 256 + oo] = acc_pred[v] * (1.0f / (float)T_STEPS) + bias;
    }
  }
}

// ---- host entry -------------------------------------------------------------
extern "C" void kernel_launch(void* const* d_in, const int* in_sizes, int n_in,
                              void* d_out, int out_size, void* d_ws, size_t ws_size,
                              hipStream_t stream) {
  (void)in_sizes; (void)n_in; (void)out_size; (void)ws_size;
  const float* seq       = (const float*)d_in[0];
  const float* delay     = (const float*)d_in[1];
  const float* h0_enc    = (const float*)d_in[2];
  const float* h0_mask   = (const float*)d_in[3];
  const float* h0_pred   = (const float*)d_in[4];
  const float* enc_w_ih  = (const float*)d_in[5];
  const float* enc_w_hh  = (const float*)d_in[6];
  const float* enc_b_ih  = (const float*)d_in[7];
  const float* enc_b_hh  = (const float*)d_in[8];
  const float* mdec_w_ih = (const float*)d_in[9];
  const float* mdec_w_hh = (const float*)d_in[10];
  const float* mdec_b_ih = (const float*)d_in[11];
  const float* mdec_b_hh = (const float*)d_in[12];
  const float* mlin_w    = (const float*)d_in[13];
  const float* mlin_b    = (const float*)d_in[14];
  const float* pdec_w_ih = (const float*)d_in[15];
  const float* pdec_w_hh = (const float*)d_in[16];
  const float* pdec_b_ih = (const float*)d_in[17];
  const float* pdec_b_hh = (const float*)d_in[18];
  const float* plin_w    = (const float*)d_in[19];
  const float* plin_b    = (const float*)d_in[20];

  char* ws = (char*)d_ws;
  unsigned* sync_cnt = (unsigned*)ws;                 // [0]=counter, [1]=generation
  unsigned* sync_gen = sync_cnt + 1;
  unsigned short* He = (unsigned short*)(ws + 256);   // bf16 states, double buffered
  unsigned short* Hm = He + 2 * NSTATE;
  unsigned short* Hp = Hm + 2 * NSTATE;
  unsigned short* Wenc = Hp + 2 * NSTATE;             // packed bf16 weight slabs
  unsigned short* Wmih = Wenc + 768 * HID;
  unsigned short* Wmhh = Wmih + 768 * HID;
  unsigned short* Wpih = Wmhh + 768 * HID;
  unsigned short* Wphh = Wpih + 768 * HID;
  unsigned short* Wpl  = Wphh + 768 * HID;

  float* mask_out = (float*)d_out;                    // [T,B,1]
  float* pred_out = mask_out + (size_t)T_STEPS * BATCH;  // [B,P]

  hipMemsetAsync(d_ws, 0, 256, stream);               // barrier state: deterministic per call
  pack_weights<<<48, 256, 0, stream>>>(enc_w_hh,  Wenc, 768);
  pack_weights<<<48, 256, 0, stream>>>(mdec_w_ih, Wmih, 768);
  pack_weights<<<48, 256, 0, stream>>>(mdec_w_hh, Wmhh, 768);
  pack_weights<<<48, 256, 0, stream>>>(pdec_w_ih, Wpih, 768);
  pack_weights<<<48, 256, 0, stream>>>(pdec_w_hh, Wphh, 768);
  pack_weights<<<16, 256, 0, stream>>>(plin_w,    Wpl,  256);
  init_states<<<NSTATE / 256, 256, 0, stream>>>(h0_enc, h0_mask, h0_pred, He, Hm, Hp);
  init_mask<<<(T_STEPS * BATCH) / 256, 256, 0, stream>>>(mask_out, mlin_b);

  rnn_persistent<<<NBLOCKS, 256, 0, stream>>>(
      seq, delay,
      enc_w_ih, enc_b_ih, enc_b_hh,
      mdec_b_ih, mdec_b_hh, mlin_w,
      pdec_b_ih, pdec_b_hh, plin_b,
      h0_enc, h0_mask, h0_pred,
      Wenc, Wmih, Wmhh, Wpih, Wphh, Wpl,
      He, Hm, Hp, mask_out, pred_out,
      sync_cnt, sync_gen);
}
